// AttentionHeadGATV2_24953759989860
// MI455X (gfx1250) — compile-verified, matrix operand order checked by reading, not run on hardware
//
#include <hip/hip_runtime.h>
#include <hip/hip_bf16.h>

typedef __attribute__((ext_vector_type(16))) _Float16 v16h;
typedef __attribute__((ext_vector_type(4)))  _Float16 h4;
typedef __attribute__((ext_vector_type(8)))  float    v8f;

#define FD 128   // feature dim == units

__device__ __forceinline__ float leaky_f(float v) { return v > 0.0f ? v : 0.2f * v; }

// order-preserving float <-> uint key (for atomic max over signed floats)
__device__ __forceinline__ unsigned fkey(float f) {
    unsigned b = __float_as_uint(f);
    return (b & 0x80000000u) ? ~b : (b | 0x80000000u);
}
__device__ __forceinline__ float funkey(unsigned k) {
    unsigned b = (k & 0x80000000u) ? (k & 0x7fffffffu) : ~k;
    return __uint_as_float(b);
}

// WMMA 16-bit fragment layout for a K=32 block (A 16x32 and B 32x16 share it):
//   element kk (0..31) of row/col m lives at lane = m + 16*kk[3],
//   half-slot = 2*((kk[4]<<2) | kk[2:1]) + kk[0]
__device__ __forceinline__ int frag_lane_off(int kk) { return ((kk >> 3) & 1) * 16; }
__device__ __forceinline__ int frag_slot(int kk) {
    return 2 * ((((kk >> 4) & 1) << 2) | ((kk >> 1) & 3)) + (kk & 1);
}

// ---------------- Phase 0: zero accumulators ----------------
__global__ void zero_kernel(float* __restrict__ out_acc, float* __restrict__ denom,
                            unsigned* __restrict__ amax_key, int n_nodes) {
    int idx = blockIdx.x * blockDim.x + threadIdx.x;
    if (idx < n_nodes * FD) out_acc[idx] = 0.0f;
    if (idx < n_nodes) { denom[idx] = 0.0f; amax_key[idx] = 0u; }  // key 0 ~= -inf
}

// ---------------- Phase 1: fused node-projection GEMM via WMMA ----------------
// Computes three 128-wide panels of node @ [W_lin | W_att_top | W_att_bot]:
//   w_n = node@W_lin + b_lin ; p_r = node@W_att[:128] + b_att ; p_s = node@W_att[128:]
// grid: (n_nodes/16, 6), block 128 (4 waves). Wave w of block (x,y) owns the
// 16x16 C-tile at rows 16*x, panel cols (y*64 + w*16).
// LDS tiles are staged PRE-PERMUTED into WMMA fragment order so each lane loads
// its whole 16-half fragment as one aligned 32-byte vector.
__global__ void __launch_bounds__(128)
proj_gemm_kernel(const float* __restrict__ node,
                 const float* __restrict__ W_lin, const float* __restrict__ b_lin,
                 const float* __restrict__ W_att, const float* __restrict__ b_att,
                 float* __restrict__ w_n, float* __restrict__ p_r, float* __restrict__ p_s,
                 int n_nodes) {
    // Ashf[kb][lane][slot] : 4*32*16 halves = 4 KB
    // Bshf[kb][wave][lane][slot] : 4*4*32*16 halves = 16 KB
    __shared__ alignas(32) _Float16 Ashf[4 * 32 * 16];
    __shared__ alignas(32) _Float16 Bshf[4 * 4 * 32 * 16];

    const int row0 = blockIdx.x * 16;
    const int lane = threadIdx.x & 31;
    const int wave = threadIdx.x >> 5;
    const bool full_tile = (row0 + 16 <= n_nodes);   // wave-uniform

    const int py    = blockIdx.y;      // 0..5 -> which 64-col chunk of the 384 output cols
    const int panel = py >> 1;         // 0: w_n, 1: p_r, 2: p_s
    const int cbase = (py & 1) * 64;   // column base within the 128-wide panel
    const float* Wsrc = (panel == 0) ? W_lin : (panel == 1 ? W_att : W_att + 128 * FD);
    const float* bias = (panel == 0) ? b_lin : (panel == 1 ? b_att : nullptr);
    float*       dst  = (panel == 0) ? w_n   : (panel == 1 ? p_r   : p_s);

    // ---- stage A tile (16 rows x 128 K) in fragment order: 512 float4 chunks ----
    #pragma unroll
    for (int it = 0; it < 4; ++it) {
        int chunk = threadIdx.x + it * 128;        // 0..511
        int r  = chunk >> 5;                        // node row within tile
        int c4 = (chunk & 31) * 4;                  // K, 4-aligned
        int row = row0 + r;
        float4 f = make_float4(0.f, 0.f, 0.f, 0.f);
        if (full_tile || row < n_nodes) f = *(const float4*)(node + (size_t)row * FD + c4);
        int kb = c4 >> 5, kk = c4 & 31;
        int ln = r + frag_lane_off(kk);
        int s  = frag_slot(kk);                     // 4-aligned for 4-aligned kk
        h4 hv; hv[0] = (_Float16)f.x; hv[1] = (_Float16)f.y;
               hv[2] = (_Float16)f.z; hv[3] = (_Float16)f.w;
        *(h4*)(Ashf + ((kb * 32 + ln) * 16 + s)) = hv;
    }

    // ---- stage B chunk (128 K x 64 cols) in fragment order: 2048 K-quads ----
    #pragma unroll
    for (int it = 0; it < 16; ++it) {
        int task = threadIdx.x + it * 128;          // 0..2047
        int nl  = task & 63;                        // column within 64-col chunk
        int k4  = (task >> 6) * 4;                  // K, 4-aligned
        const float* wp = Wsrc + (size_t)k4 * FD + cbase + nl;
        float w0 = wp[0 * FD], w1 = wp[1 * FD], w2 = wp[2 * FD], w3 = wp[3 * FD];
        int kb = k4 >> 5, kk = k4 & 31;
        int ln = (nl & 15) + frag_lane_off(kk);
        int wv = nl >> 4;                           // which wave's tile
        int s  = frag_slot(kk);
        h4 hv; hv[0] = (_Float16)w0; hv[1] = (_Float16)w1;
               hv[2] = (_Float16)w2; hv[3] = (_Float16)w3;
        *(h4*)(Bshf + (((kb * 4 + wv) * 32 + ln) * 16 + s)) = hv;
    }
    __syncthreads();

    v8f acc = {};
    #pragma unroll
    for (int kb = 0; kb < 4; ++kb) {
        v16h af = *(const v16h*)(Ashf + (kb * 32 + lane) * 16);
        v16h bf = *(const v16h*)(Bshf + ((kb * 4 + wave) * 32 + lane) * 16);
        acc = __builtin_amdgcn_wmma_f32_16x16x32_f16(
            /*neg_a=*/false, af, /*neg_b=*/false, bf,
            /*c_mod=*/(short)0, acc, /*reuse_a=*/false, /*reuse_b=*/false);
    }

    // C/D layout: VGPR r -> row r (lanes 0-15) or row 8+r (lanes 16-31), col = lane&15
    const int cB = cbase + wave * 16 + (lane & 15); // column within the 128-wide panel
    const int rb = (lane >> 4) * 8;
    float bv = 0.0f;
    if (bias) bv = bias[cB];                        // uniform branch (panel-dependent)
    float* outp = dst + (size_t)(row0 + rb) * FD + cB;
    if (full_tile) {                                // uniform: no per-element exec masking
        #pragma unroll
        for (int r = 0; r < 8; ++r) outp[(size_t)r * FD] = acc[r] + bv;
    } else {
        #pragma unroll
        for (int r = 0; r < 8; ++r)
            if (row0 + rb + r < n_nodes) outp[(size_t)r * FD] = acc[r] + bv;
    }
}

// ---------------- Phase 2: per-edge attention logits + segment max ----------------
// one wave32 per edge; lanes each take 4 of the 128 dims (float4)
__global__ void __launch_bounds__(256)
edge_logits_kernel(const int* __restrict__ eidx, const float* __restrict__ p_r,
                   const float* __restrict__ p_s, const float* __restrict__ a_vec,
                   float* __restrict__ elog, unsigned* __restrict__ amax_key, int n_edges) {
    const int lane = threadIdx.x & 31;
    const int e = blockIdx.x * 8 + (threadIdx.x >> 5);
    if (e >= n_edges) return;
    const int i = eidx[e];              // receiver
    const int j = eidx[n_edges + e];    // sender
    const float4 a  = ((const float4*)(p_r + (size_t)i * FD))[lane];
    const float4 b  = ((const float4*)(p_s + (size_t)j * FD))[lane];
    const float4 av = ((const float4*)a_vec)[lane];
    float s = leaky_f(a.x + b.x) * av.x + leaky_f(a.y + b.y) * av.y
            + leaky_f(a.z + b.z) * av.z + leaky_f(a.w + b.w) * av.w;
    #pragma unroll
    for (int off = 16; off > 0; off >>= 1) s += __shfl_xor(s, off, 32);
    if (lane == 0) {
        elog[e] = s;
        atomicMax(&amax_key[i], fkey(s));
    }
}

// ---------------- Phase 3: exp(max-shifted) + segment sum ----------------
__global__ void edge_exp_kernel(const int* __restrict__ eidx, const unsigned* __restrict__ amax_key,
                                float* __restrict__ elog, float* __restrict__ denom, int n_edges) {
    int e = blockIdx.x * blockDim.x + threadIdx.x;
    if (e >= n_edges) return;
    int i = eidx[e];
    float ex = expf(elog[e] - funkey(amax_key[i]));
    elog[e] = ex;                       // overwrite logit with exp (own slot, safe)
    atomicAdd(&denom[i], ex);
}

// ---------------- Phase 4: attention-weighted scatter of messages ----------------
// one wave32 per edge; float4 per lane; f32 atomics into receiver row
__global__ void __launch_bounds__(256)
edge_scatter_kernel(const int* __restrict__ eidx, const float* __restrict__ elog,
                    const float* __restrict__ denom, const float* __restrict__ w_n,
                    float* __restrict__ out_acc, int n_edges) {
    const int lane = threadIdx.x & 31;
    const int e = blockIdx.x * 8 + (threadIdx.x >> 5);
    if (e >= n_edges) return;
    const int i = eidx[e];
    const int j = eidx[n_edges + e];
    const float alpha = elog[e] / denom[i];
    const float4 v = ((const float4*)(w_n + (size_t)j * FD))[lane];
    float* dst = out_acc + (size_t)i * FD + lane * 4;
    atomicAdd(dst + 0, alpha * v.x);
    atomicAdd(dst + 1, alpha * v.y);
    atomicAdd(dst + 2, alpha * v.z);
    atomicAdd(dst + 3, alpha * v.w);
}

// ---------------- Phase 5: final leaky activation (in place) ----------------
__global__ void final_leaky_kernel(float* __restrict__ out, int total) {
    int idx = blockIdx.x * blockDim.x + threadIdx.x;
    if (idx < total) out[idx] = leaky_f(out[idx]);
}

extern "C" void kernel_launch(void* const* d_in, const int* in_sizes, int n_in,
                              void* d_out, int out_size, void* d_ws, size_t ws_size,
                              hipStream_t stream) {
    const float* node  = (const float*)d_in[0];
    // d_in[1] = edge features: unused by the reference math (use_edge_features=False)
    const int*   eidx  = (const int*)d_in[2];
    const float* W_lin = (const float*)d_in[3];
    const float* b_lin = (const float*)d_in[4];
    const float* W_att = (const float*)d_in[5];
    const float* b_att = (const float*)d_in[6];
    const float* a_vec = (const float*)d_in[7];
    float* out = (float*)d_out;

    const int n_nodes = in_sizes[0] / FD;
    const int n_edges = in_sizes[2] / 2;

    // workspace carve-out (~80 MB): w_n, p_r, p_s [N,128]; elog [M]; denom, amax_key [N]
    char* ws = (char*)d_ws;
    size_t off = 0;
    auto carve = [&](size_t bytes) -> char* {
        char* p = ws + off;
        off = (off + bytes + 255) & ~(size_t)255;
        return p;
    };
    float*    w_n      = (float*)carve((size_t)n_nodes * FD * sizeof(float));
    float*    p_r      = (float*)carve((size_t)n_nodes * FD * sizeof(float));
    float*    p_s      = (float*)carve((size_t)n_nodes * FD * sizeof(float));
    float*    elog     = (float*)carve((size_t)n_edges * sizeof(float));
    float*    denom    = (float*)carve((size_t)n_nodes * sizeof(float));
    unsigned* amax_key = (unsigned*)carve((size_t)n_nodes * sizeof(unsigned));
    (void)ws_size; (void)n_in; (void)out_size;

    const int total = n_nodes * FD;
    zero_kernel<<<(total + 255) / 256, 256, 0, stream>>>(out, denom, amax_key, n_nodes);

    dim3 ggrid((n_nodes + 15) / 16, 6);  // 6 * 4 waves = 24 tiles of 16 cols = 384 output cols
    proj_gemm_kernel<<<ggrid, 128, 0, stream>>>(node, W_lin, b_lin, W_att, b_att,
                                                w_n, p_r, p_s, n_nodes);

    const int eblocks = (n_edges + 7) / 8;  // 8 waves (edges) per 256-thread block
    edge_logits_kernel<<<eblocks, 256, 0, stream>>>(eidx, p_r, p_s, a_vec, elog, amax_key, n_edges);
    edge_exp_kernel<<<(n_edges + 255) / 256, 256, 0, stream>>>(eidx, amax_key, elog, denom, n_edges);
    edge_scatter_kernel<<<eblocks, 256, 0, stream>>>(eidx, elog, denom, w_n, out, n_edges);
    final_leaky_kernel<<<(total + 255) / 256, 256, 0, stream>>>(out, total);
}
